// GptInferenceModel_85950885527629
// MI455X (gfx1250) — matrix operand order repaired and checked by reading, not compiled, not run on hardware
//
#include <hip/hip_runtime.h>
#include <hip/hip_bf16.h>
#include <stdint.h>

// ---- model constants -------------------------------------------------------
#define V_    50257
#define D_    512
#define H_    8
#define F_    2048
#define L_    4
#define DH_   64
#define L0_   512
#define STEPS_ 32
#define TOPK_ 10
#define T_    544     // L0 + STEPS
#define B_    8

typedef _Float16 f16;
typedef __attribute__((ext_vector_type(16))) _Float16 v16h;
typedef __attribute__((ext_vector_type(8)))  float    v8f;

union AFrag { v16h v; f16 h[16]; unsigned u[8]; };
union Pk    { unsigned u; f16 h[2]; };

static __device__ __forceinline__ v8f wmma_f16(const AFrag& a, const AFrag& b, v8f c) {
  // (neg_a, A, neg_b, B, c_mod, C, reuse_a, reuse_b)
  return __builtin_amdgcn_wmma_f32_16x16x32_f16(false, a.v, false, b.v, (short)0, c, false, false);
}

// ---- weight packing into WMMA B-fragment layout ----------------------------
// dst[(((kt*(Np/16) + nt)*32 + lane)*16 + j)] = W[kt*32 + (lane>>4)*16 + j][nt*16 + (lane&15)]
// (zero-padded for n >= N; transpose=1 reads src[n*K + k] instead of src[k*N + n])
__global__ void pack_w_kernel(const float* __restrict__ src, f16* __restrict__ dst,
                              int K, int N, int Np, int transpose) {
  long i = (long)blockIdx.x * blockDim.x + threadIdx.x;
  long total = (long)K * Np;
  if (i >= total) return;
  int j    = (int)(i & 15);
  int lane = (int)((i >> 4) & 31);
  long rest = i >> 9;
  int ntiles = Np >> 4;
  int nt = (int)(rest % ntiles);
  int kt = (int)(rest / ntiles);
  int n = nt * 16 + (lane & 15);
  int k = kt * 32 + (lane >> 4) * 16 + j;
  float v = 0.f;
  if (n < N) v = transpose ? src[(long)n * K + k] : src[(long)k * N + n];
  dst[i] = (f16)v;
}

// ---- embedding -------------------------------------------------------------
__global__ void copy_tokens_kernel(const int* __restrict__ tok, int* __restrict__ tbuf) {
  int i = blockIdx.x * blockDim.x + threadIdx.x;
  if (i < B_ * L0_) {
    int b = i / L0_, t = i % L0_;
    tbuf[b * T_ + t] = tok[i];
  }
}

__global__ void embed_prefill_kernel(const int* __restrict__ tok, const float* __restrict__ emb,
                                     const float* __restrict__ pos, const float* __restrict__ depth,
                                     float* __restrict__ x0) {
  long i = (long)blockIdx.x * blockDim.x + threadIdx.x;
  if (i < (long)B_ * L0_ * D_) {
    int d = (int)(i % D_);
    long r = i / D_;
    int t = (int)(r % L0_);
    int b = (int)(r / L0_);
    int tk = tok[b * L0_ + t];
    x0[((long)b * T_ + t) * D_ + d] = emb[(long)tk * D_ + d] + pos[(long)t * D_ + d] + depth[d];
  }
}

__global__ void embed_decode_kernel(const int* __restrict__ tbuf, const float* __restrict__ emb,
                                    const float* __restrict__ pos, const float* __restrict__ depth,
                                    float* __restrict__ x0, int p) {
  int i = blockIdx.x * blockDim.x + threadIdx.x;
  if (i < B_ * D_) {
    int d = i % D_, b = i / D_;
    int tk = tbuf[b * T_ + p];
    x0[((long)b * T_ + p) * D_ + d] = emb[(long)tk * D_ + d] + pos[(long)p * D_ + d] + depth[d];
  }
}

// ---- generic WMMA GEMM: out = act(A[M,K] * Wpacked[K,Np] + bias (+res)) ----
// Row r -> (b = r/rows_per_b, t = r%rows_per_b); A row at A + b*a_b + t*K,
// out row at out + b*o_b + t*N, res row at res + b*r_b + t*N.
// Each wave computes a 32x32 output tile (2 A-frags x 2 B-frags, 4 WMMAs/k-step);
// block = 8 waves = 32 x 256 output. A tile double-buffered in LDS (1 barrier/iter).
__global__ __launch_bounds__(256) void gemm_wmma_kernel(
    const float* __restrict__ A, const f16* __restrict__ Wp,
    const float* __restrict__ bias, const float* __restrict__ res,
    float* __restrict__ out,
    int M, int N, int Np, int K, int rows_per_b,
    long a_b, long o_b, long r_b, int relu)
{
  __shared__ unsigned As[2][32][17];          // double-buffered 32 rows x 32 halves
  const int tid    = threadIdx.x;
  const int wave   = tid >> 5;
  const int lane   = tid & 31;
  const int halfid = lane >> 4;
  const int ln16   = lane & 15;
  const int ntiles = Np >> 4;
  const int m0 = blockIdx.y * 32;

  int ntA = blockIdx.x * 16 + wave * 2;
  int ntB = ntA + 1;
  if (ntA > ntiles - 1) ntA = ntiles - 1;
  if (ntB > ntiles - 1) ntB = ntiles - 1;

  // cooperative A loader: thread -> rows (tid>>4) and (tid>>4)+16, pair tid&15
  const int lrow  = tid >> 4;
  const int lpair = tid & 15;
  auto rowptr = [&](int gr) -> const float* {
    if (gr > M - 1) gr = M - 1;
    int bb = gr / rows_per_b, tt = gr % rows_per_b;
    return A + (long)bb * a_b + (long)tt * K;
  };
  const float* arow0 = rowptr(m0 + lrow);
  const float* arow1 = rowptr(m0 + lrow + 16);

  // preload k-tile 0 into buffer 0
  {
    Pk p0, p1;
    p0.h[0] = (f16)arow0[2 * lpair]; p0.h[1] = (f16)arow0[2 * lpair + 1];
    p1.h[0] = (f16)arow1[2 * lpair]; p1.h[1] = (f16)arow1[2 * lpair + 1];
    As[0][lrow][lpair]      = p0.u;
    As[0][16 + lrow][lpair] = p1.u;
  }
  __syncthreads();

  v8f c00 = {}, c01 = {}, c10 = {}, c11 = {};
  const int KT = K >> 5;
  for (int kt = 0; kt < KT; ++kt) {
    const int buf = kt & 1;
    // issue next A-tile global loads early (clamped on last iter; store skipped)
    Pk n0, n1;
    {
      const int kn = ((kt + 1 < KT) ? kt + 1 : kt) << 5;
      n0.h[0] = (f16)arow0[kn + 2 * lpair]; n0.h[1] = (f16)arow0[kn + 2 * lpair + 1];
      n1.h[0] = (f16)arow1[kn + 2 * lpair]; n1.h[1] = (f16)arow1[kn + 2 * lpair + 1];
    }
    // B fragments (packed, contiguous 32B per lane)
    AFrag b0, b1;
    {
      const long base = ((long)kt * ntiles) * 512 + (long)lane * 16;
      b0.v = *(const v16h*)(Wp + base + (long)ntA * 512);
      b1.v = *(const v16h*)(Wp + base + (long)ntB * 512);
      __builtin_prefetch(Wp + base + ((long)ntiles + (long)ntA) * 512, 0, 1);
    }
    // A fragments from LDS
    AFrag a0, a1;
    {
      const int pb = halfid * 8;
      #pragma unroll
      for (int i = 0; i < 4; ++i) {
        a0.u[i]     = As[buf][ln16][pb + i];
        a0.u[4 + i] = As[buf][ln16][pb + 4 + i];
        a1.u[i]     = As[buf][16 + ln16][pb + i];
        a1.u[4 + i] = As[buf][16 + ln16][pb + 4 + i];
      }
    }
    c00 = wmma_f16(a0, b0, c00);
    c01 = wmma_f16(a0, b1, c01);
    c10 = wmma_f16(a1, b0, c10);
    c11 = wmma_f16(a1, b1, c11);

    if (kt + 1 < KT) {
      As[buf ^ 1][lrow][lpair]      = n0.u;
      As[buf ^ 1][16 + lrow][lpair] = n1.u;
    }
    __syncthreads();
  }

  #pragma unroll
  for (int g = 0; g < 2; ++g) {
    #pragma unroll
    for (int t2 = 0; t2 < 2; ++t2) {
      const int n = (t2 ? ntB : ntA) * 16 + ln16;
      const v8f& c = g ? (t2 ? c11 : c10) : (t2 ? c01 : c00);
      if (n < N) {
        const float bi = bias ? bias[n] : 0.f;
        #pragma unroll
        for (int v = 0; v < 8; ++v) {
          int m = m0 + g * 16 + v + 8 * halfid;
          if (m < M) {
            int bb = m / rows_per_b, tt = m % rows_per_b;
            float val = c[v] + bi;
            if (res)  val += res[(long)bb * r_b + (long)tt * N + n];
            if (relu) val = fmaxf(val, 0.f);
            out[(long)bb * o_b + (long)tt * N + n] = val;
          }
        }
      }
    }
  }
}

// ---- attention: one wave per (q-tile, head, batch) -------------------------
// Q rows: Q + b*q_b + r*D + h*DH (r = qt*16 + m, valid if r < n_valid)
// keys j in [0, kvlen): K/V at base + b*kv_b + j*D + h*DH
// mask: score(m, j) valid iff j <= qpos0 + qt*16 + m
__global__ __launch_bounds__(32) void attn_wmma_kernel(
    const float* __restrict__ Q, const float* __restrict__ Kb,
    const float* __restrict__ Vb, float* __restrict__ O,
    long q_b, long kv_b, long o_b,
    int qpos0, int n_valid, int kvlen, float scale)
{
  __shared__ float S[16][T_];
  __shared__ f16   P[16][T_];
  const int b = blockIdx.z, h = blockIdx.y, qt = blockIdx.x;
  const int lane   = threadIdx.x & 31;
  const int halfid = lane >> 4;
  const int ln16   = lane & 15;
  const int kb     = halfid * 8;
  const int kk     = halfid * 16;

  // Q fragments for k 0..31 and 32..63
  AFrag aq0, aq1;
  {
    int qr = qt * 16 + ln16;
    if (qr >= n_valid) qr = n_valid - 1;
    const float* qp = Q + (long)b * q_b + (long)qr * D_ + h * DH_;
    #pragma unroll
    for (int j = 0; j < 16; ++j) {
      int kid = kk + j;
      aq0.h[j] = (f16)qp[kid];
      aq1.h[j] = (f16)qp[32 + kid];
    }
  }

  // pass 1: S = scale * Q*K^T (clamped key addresses; invalid cols masked below)
  for (int j0 = 0; j0 < kvlen; j0 += 16) {
    int key = j0 + ln16;
    if (key > kvlen - 1) key = kvlen - 1;
    const float* kp = Kb + (long)b * kv_b + (long)key * D_ + h * DH_;
    AFrag bk0, bk1;
    #pragma unroll
    for (int j = 0; j < 16; ++j) {         // 16 consecutive f32 -> wide loads
      bk0.h[j] = (f16)kp[kk + j];
      bk1.h[j] = (f16)kp[32 + kk + j];
    }
    v8f s = {};
    s = wmma_f16(aq0, bk0, s);
    s = wmma_f16(aq1, bk1, s);
    #pragma unroll
    for (int v = 0; v < 8; ++v) {
      int m = v + 8 * halfid;
      int qpos = qpos0 + qt * 16 + m;
      int j = j0 + ln16;
      float val = s[v] * scale;
      if (j > qpos || j >= kvlen) val = -1e9f;
      S[m][j] = val;
    }
  }
  __syncthreads();

  // pass 2: row softmax -> P (f16), zero-padded to multiple of 32
  const int kpad = (kvlen + 31) & ~31;
  for (int r = 0; r < 16; ++r) {
    float mx = -3.4e38f;
    for (int j = lane; j < kvlen; j += 32) mx = fmaxf(mx, S[r][j]);
    #pragma unroll
    for (int off = 16; off >= 1; off >>= 1) mx = fmaxf(mx, __shfl_xor(mx, off, 32));
    float sm = 0.f;
    for (int j = lane; j < kvlen; j += 32) {
      float e = __expf(S[r][j] - mx);
      S[r][j] = e;
      sm += e;
    }
    #pragma unroll
    for (int off = 16; off >= 1; off >>= 1) sm += __shfl_xor(sm, off, 32);
    float inv = 1.f / sm;
    for (int j = lane; j < kpad; j += 32)
      P[r][j] = (j < kvlen) ? (f16)(S[r][j] * inv) : (f16)0.f;
  }
  __syncthreads();

  // pass 3: O = P * V  (4 column tiles of DH; P==0 beyond kvlen kills clamped V)
  v8f co[4]; co[0] = {}; co[1] = {}; co[2] = {}; co[3] = {};
  for (int k0 = 0; k0 < kpad; k0 += 32) {
    AFrag ap;
    #pragma unroll
    for (int j = 0; j < 16; ++j) {
      int kid = kk + j;
      ap.h[j] = P[ln16][k0 + kid];
    }
    #pragma unroll
    for (int nt = 0; nt < 4; ++nt) {
      AFrag bv;
      const int dcol = h * DH_ + nt * 16 + ln16;
      #pragma unroll
      for (int i = 0; i < 16; ++i) {
        int key = k0 + kk + i;
        if (key > kvlen - 1) key = kvlen - 1;
        bv.h[i] = (f16)Vb[(long)b * kv_b + (long)key * D_ + dcol];
      }
      co[nt] = wmma_f16(ap, bv, co[nt]);
    }
  }
  #pragma unroll
  for (int nt = 0; nt < 4; ++nt) {
    #pragma unroll
    for (int v = 0; v < 8; ++v) {
      int m = v + 8 * halfid;
      int r = qt * 16 + m;
      if (r < n_valid)
        O[(long)b * o_b + (long)r * D_ + h * DH_ + nt * 16 + ln16] = co[nt][v];
    }
  }
}

// ---- layernorm -------------------------------------------------------------
__global__ __launch_bounds__(256) void ln_kernel(
    const float* __restrict__ in, long in_b,
    const float* __restrict__ g, const float* __restrict__ bt,
    float* __restrict__ out, long out_b, int rows_per_b)
{
  __shared__ float red[256];
  int r = blockIdx.x;
  int bb = r / rows_per_b, tt = r % rows_per_b;
  const float* x = in + (long)bb * in_b + (long)tt * D_;
  float s = 0.f;
  for (int d = threadIdx.x; d < D_; d += 256) s += x[d];
  red[threadIdx.x] = s; __syncthreads();
  for (int off = 128; off >= 1; off >>= 1) {
    if (threadIdx.x < off) red[threadIdx.x] += red[threadIdx.x + off];
    __syncthreads();
  }
  float mu = red[0] / (float)D_;
  __syncthreads();
  s = 0.f;
  for (int d = threadIdx.x; d < D_; d += 256) { float t = x[d] - mu; s += t * t; }
  red[threadIdx.x] = s; __syncthreads();
  for (int off = 128; off >= 1; off >>= 1) {
    if (threadIdx.x < off) red[threadIdx.x] += red[threadIdx.x + off];
    __syncthreads();
  }
  float inv = rsqrtf(red[0] / (float)D_ + 1e-5f);
  float* o = out + (long)bb * out_b + (long)tt * D_;
  for (int d = threadIdx.x; d < D_; d += 256)
    o[d] = (x[d] - mu) * inv * g[d] + bt[d];
}

// ---- top-k + categorical sample -------------------------------------------
__global__ __launch_bounds__(256) void sample_kernel(
    float* __restrict__ logits, float* __restrict__ out_probs,
    int* __restrict__ tbuf, int step)
{
  __shared__ float sv[256]; __shared__ int si[256];
  __shared__ float kvs[TOPK_]; __shared__ int kis[TOPK_];
  const int b = blockIdx.x;
  float* lg = logits + (long)b * V_;

  // softmax stats over full vocab
  float mx = -3.4e38f;
  for (int i = threadIdx.x; i < V_; i += 256) mx = fmaxf(mx, lg[i]);
  sv[threadIdx.x] = mx; __syncthreads();
  for (int off = 128; off >= 1; off >>= 1) {
    if (threadIdx.x < off) sv[threadIdx.x] = fmaxf(sv[threadIdx.x], sv[threadIdx.x + off]);
    __syncthreads();
  }
  mx = sv[0]; __syncthreads();
  float sm = 0.f;
  for (int i = threadIdx.x; i < V_; i += 256) sm += __expf(lg[i] - mx);
  sv[threadIdx.x] = sm; __syncthreads();
  for (int off = 128; off >= 1; off >>= 1) {
    if (threadIdx.x < off) sv[threadIdx.x] += sv[threadIdx.x + off];
    __syncthreads();
  }
  sm = sv[0]; __syncthreads();

  // top-k by iterative argmax (mutates logits scratch)
  for (int it = 0; it < TOPK_; ++it) {
    float bvv = -3.4e38f; int bii = 0;
    for (int i = threadIdx.x; i < V_; i += 256) {
      float v = lg[i];
      if (v > bvv) { bvv = v; bii = i; }
    }
    sv[threadIdx.x] = bvv; si[threadIdx.x] = bii; __syncthreads();
    for (int off = 128; off >= 1; off >>= 1) {
      if (threadIdx.x < off) {
        float vo = sv[threadIdx.x + off]; int io = si[threadIdx.x + off];
        if (vo > sv[threadIdx.x] || (vo == sv[threadIdx.x] && io < si[threadIdx.x])) {
          sv[threadIdx.x] = vo; si[threadIdx.x] = io;
        }
      }
      __syncthreads();
    }
    if (threadIdx.x == 0) { kvs[it] = sv[0]; kis[it] = si[0]; lg[si[0]] = -3.4e38f; }
    __syncthreads();
  }

  if (threadIdx.x == 0) {
    float pv[TOPK_], p2[TOPK_];
    float mx2 = -3.4e38f;
    for (int i = 0; i < TOPK_; ++i) { pv[i] = __expf(kvs[i] - mx) / sm; mx2 = fmaxf(mx2, pv[i]); }
    float s2 = 0.f;
    for (int i = 0; i < TOPK_; ++i) { p2[i] = __expf(pv[i] - mx2); s2 += p2[i]; }
    unsigned xr = 0x9E3779B9u * (unsigned)(step + 1) + 0x85EBCA6Bu * (unsigned)(b + 1) + 0x2545F491u;
    xr ^= xr << 13; xr ^= xr >> 17; xr ^= xr << 5;
    xr ^= xr << 13; xr ^= xr >> 17; xr ^= xr << 5;
    float u = (float)(xr >> 8) * (1.0f / 16777216.0f);
    float c = 0.f; int pick = TOPK_ - 1;
    for (int i = 0; i < TOPK_; ++i) { c += p2[i] / s2; if (u < c) { pick = i; break; } }
    for (int i = 0; i < TOPK_; ++i)
      out_probs[((long)step * B_ + b) * TOPK_ + i] = p2[i] / s2;
    tbuf[b * T_ + (L0_ + step)] = kis[pick];
  }
}

__global__ void tokens_out_kernel(const int* __restrict__ tbuf, float* __restrict__ out) {
  int i = blockIdx.x * blockDim.x + threadIdx.x;
  if (i < B_ * T_) out[i] = (float)tbuf[i];
}

// ---- host orchestration ----------------------------------------------------
extern "C" void kernel_launch(void* const* d_in, const int* in_sizes, int n_in,
                              void* d_out, int out_size, void* d_ws, size_t ws_size,
                              hipStream_t stream) {
  (void)in_sizes; (void)n_in; (void)out_size; (void)ws_size;
  const int*   tokens = (const int*)  d_in[0];
  const float* emb    = (const float*)d_in[1];
  const float* pos    = (const float*)d_in[2];
  const float* depth  = (const float*)d_in[3];
  const float* wq = (const float*)d_in[4];
  const float* bq = (const float*)d_in[5];
  const float* wk = (const float*)d_in[6];
  const float* bk = (const float*)d_in[7];
  const float* wv = (const float*)d_in[8];
  const float* bv = (const float*)d_in[9];
  const float* wo = (const float*)d_in[10];
  const float* bo = (const float*)d_in[11];
  const float* ln1g = (const float*)d_in[12];
  const float* ln1b = (const float*)d_in[13];
  const float* ln2g = (const float*)d_in[14];
  const float* ln2b = (const float*)d_in[15];
  const float* w1 = (const float*)d_in[16];
  const float* b1 = (const float*)d_in[17];
  const float* w2 = (const float*)d_in[18];
  const float* b2 = (const float*)d_in[19];
  float* out = (float*)d_out;

  const int VP = (V_ + 15) & ~15;            // 50272, padded vocab for packing

  // workspace carve-up
  char* wsp = (char*)d_ws;
  auto alloc = [&](size_t bytes) -> char* {
    char* r = wsp; wsp += (bytes + 255) & ~(size_t)255; return r;
  };
  const long BTD = (long)B_ * T_ * D_;
  f16* wq_p = (f16*)alloc(sizeof(f16) * (size_t)L_ * D_ * D_);
  f16* wk_p = (f16*)alloc(sizeof(f16) * (size_t)L_ * D_ * D_);
  f16* wv_p = (f16*)alloc(sizeof(f16) * (size_t)L_ * D_ * D_);
  f16* wo_p = (f16*)alloc(sizeof(f16) * (size_t)L_ * D_ * D_);
  f16* w1_p = (f16*)alloc(sizeof(f16) * (size_t)L_ * D_ * F_);
  f16* w2_p = (f16*)alloc(sizeof(f16) * (size_t)L_ * F_ * D_);
  f16* emb_p = (f16*)alloc(sizeof(f16) * (size_t)D_ * VP);
  float* cache  = (float*)alloc(sizeof(float) * (size_t)(L_ + 1) * BTD);
  float* qb  = (float*)alloc(sizeof(float) * (size_t)BTD);
  float* kbf = (float*)alloc(sizeof(float) * (size_t)BTD);
  float* vbf = (float*)alloc(sizeof(float) * (size_t)BTD);
  float* ob  = (float*)alloc(sizeof(float) * (size_t)BTD);
  float* ub  = (float*)alloc(sizeof(float) * (size_t)BTD);
  float* xb  = (float*)alloc(sizeof(float) * (size_t)BTD);
  float* hb  = (float*)alloc(sizeof(float) * (size_t)B_ * L0_ * F_);
  float* logits = (float*)alloc(sizeof(float) * (size_t)B_ * V_);
  float* qsm = (float*)alloc(sizeof(float) * (size_t)B_ * D_);
  float* osm = (float*)alloc(sizeof(float) * (size_t)B_ * D_);
  float* usm = (float*)alloc(sizeof(float) * (size_t)B_ * D_);
  float* xsm = (float*)alloc(sizeof(float) * (size_t)B_ * D_);
  float* hsm = (float*)alloc(sizeof(float) * (size_t)B_ * F_);
  int*   tbuf = (int*)alloc(sizeof(int) * (size_t)B_ * T_);

  const float scale = 0.125f;   // 1/sqrt(64)
  const long TD = (long)T_ * D_;

  // one-time weight packing into WMMA fragment layout (f16)
  auto pack = [&](const float* src, f16* dst, int K, int N, int Np, int tr) {
    long n = (long)K * Np;
    pack_w_kernel<<<(int)((n + 255) / 256), 256, 0, stream>>>(src, dst, K, N, Np, tr);
  };
  for (int i = 0; i < L_; ++i) {
    pack(wq + (long)i * D_ * D_, wq_p + (long)i * D_ * D_, D_, D_, D_, 0);
    pack(wk + (long)i * D_ * D_, wk_p + (long)i * D_ * D_, D_, D_, D_, 0);
    pack(wv + (long)i * D_ * D_, wv_p + (long)i * D_ * D_, D_, D_, D_, 0);
    pack(wo + (long)i * D_ * D_, wo_p + (long)i * D_ * D_, D_, D_, D_, 0);
    pack(w1 + (long)i * D_ * F_, w1_p + (long)i * D_ * F_, D_, F_, F_, 0);
    pack(w2 + (long)i * F_ * D_, w2_p + (long)i * F_ * D_, F_, D_, D_, 0);
  }
  pack(emb, emb_p, D_, V_, VP, 1);           // logits B-matrix = emb^T

  copy_tokens_kernel<<<(B_ * L0_ + 255) / 256, 256, 0, stream>>>(tokens, tbuf);
  embed_prefill_kernel<<<(int)(((long)B_ * L0_ * D_ + 255) / 256), 256, 0, stream>>>(
      tokens, emb, pos, depth, cache);

  auto gemm = [&](const float* A, const f16* Wp, const float* bias, const float* res,
                  float* o, int M, int N, int K, int rows_per_b,
                  long a_b, long o_b, long r_b, int relu) {
    int Np = (N + 15) & ~15;
    dim3 g((Np / 16 + 15) / 16, (M + 31) / 32);
    gemm_wmma_kernel<<<g, 256, 0, stream>>>(A, Wp, bias, res, o,
                                            M, N, Np, K, rows_per_b, a_b, o_b, r_b, relu);
  };

  const float* cl = cache + (long)L_ * BTD;

  // ---------------- prefill ----------------
  for (int i = 0; i < L_; ++i) {
    const float* cin = cache + (long)i * BTD;
    float* cout = cache + (long)(i + 1) * BTD;
    gemm(cin, wq_p + (long)i * D_ * D_, bq + i * D_, nullptr, qb,  B_ * L0_, D_, D_, L0_, TD, TD, 0, 0);
    gemm(cin, wk_p + (long)i * D_ * D_, bk + i * D_, nullptr, kbf, B_ * L0_, D_, D_, L0_, TD, TD, 0, 0);
    gemm(cin, wv_p + (long)i * D_ * D_, bv + i * D_, nullptr, vbf, B_ * L0_, D_, D_, L0_, TD, TD, 0, 0);
    attn_wmma_kernel<<<dim3(L0_ / 16, H_, B_), 32, 0, stream>>>(
        qb, kbf, vbf, ob, TD, TD, TD, /*qpos0=*/0, /*n_valid=*/L0_, /*kvlen=*/L0_, scale);
    gemm(ob, wo_p + (long)i * D_ * D_, bo + i * D_, cin, ub, B_ * L0_, D_, D_, L0_, TD, TD, TD, 0);
    ln_kernel<<<B_ * L0_, 256, 0, stream>>>(ub, TD, ln1g + i * D_, ln1b + i * D_, xb, TD, L0_);
    gemm(xb, w1_p + (long)i * D_ * F_, b1 + i * F_, nullptr, hb, B_ * L0_, F_, D_, L0_,
         TD, (long)L0_ * F_, 0, 1);
    gemm(hb, w2_p + (long)i * F_ * D_, b2 + i * D_, xb, ub, B_ * L0_, D_, F_, L0_,
         (long)L0_ * F_, TD, TD, 0);
    ln_kernel<<<B_ * L0_, 256, 0, stream>>>(ub, TD, ln2g + i * D_, ln2b + i * D_, cout, TD, L0_);
  }

  // sample 0 (from position L0-1)
  gemm(cl + (long)(L0_ - 1) * D_, emb_p, nullptr, nullptr, logits, B_, V_, D_, 1, TD, (long)V_, 0, 0);
  sample_kernel<<<B_, 256, 0, stream>>>(logits, out + (long)B_ * T_, tbuf, 0);

  // ---------------- decode ----------------
  for (int s = 1; s < STEPS_; ++s) {
    const int pp = L0_ + s - 1;
    embed_decode_kernel<<<(B_ * D_ + 255) / 256, 256, 0, stream>>>(tbuf, emb, pos, depth, cache, pp);
    for (int i = 0; i < L_; ++i) {
      const float* cin = cache + (long)i * BTD;
      float* cout = cache + (long)(i + 1) * BTD;
      gemm(cin + (long)pp * D_, wq_p + (long)i * D_ * D_, bq + i * D_, nullptr, qsm,
           B_, D_, D_, 1, TD, (long)D_, 0, 0);
      gemm(cin, wk_p + (long)i * D_ * D_, bk + i * D_, nullptr, kbf,
           B_ * (pp + 1), D_, D_, pp + 1, TD, TD, 0, 0);
      gemm(cin, wv_p + (long)i * D_ * D_, bv + i * D_, nullptr, vbf,
           B_ * (pp + 1), D_, D_, pp + 1, TD, TD, 0, 0);
      attn_wmma_kernel<<<dim3(1, H_, B_), 32, 0, stream>>>(
          qsm, kbf, vbf, osm, (long)D_, TD, (long)D_, /*qpos0=*/pp, /*n_valid=*/1,
          /*kvlen=*/pp + 1, scale);
      gemm(osm, wo_p + (long)i * D_ * D_, bo + i * D_, cin + (long)pp * D_, usm,
           B_, D_, D_, 1, (long)D_, (long)D_, TD, 0);
      ln_kernel<<<B_, 256, 0, stream>>>(usm, (long)D_, ln1g + i * D_, ln1b + i * D_, xsm, (long)D_, 1);
      gemm(xsm, w1_p + (long)i * D_ * F_, b1 + i * F_, nullptr, hsm,
           B_, F_, D_, 1, (long)D_, (long)F_, 0, 1);
      gemm(hsm, w2_p + (long)i * F_ * D_, b2 + i * D_, xsm, usm,
           B_, D_, F_, 1, (long)F_, (long)D_, (long)D_, 0);
      ln_kernel<<<B_, 256, 0, stream>>>(usm, (long)D_, ln2g + i * D_, ln2b + i * D_,
                                        cout + (long)pp * D_, TD, 1);
    }
    gemm(cl + (long)pp * D_, emb_p, nullptr, nullptr, logits, B_, V_, D_, 1, TD, (long)V_, 0, 0);
    sample_kernel<<<B_, 256, 0, stream>>>(logits, out + (long)B_ * T_, tbuf, s);
  }

  tokens_out_kernel<<<(B_ * T_ + 255) / 256, 256, 0, stream>>>(tbuf, out);
}
